// LSTM_RNN_16527034155631
// MI455X (gfx1250) — matrix operand order repaired and checked by reading, not compiled, not run on hardware
//
#include <hip/hip_runtime.h>

// ---------------------------------------------------------------------------
// Matrix-carry LSTM on MI455X (gfx1250).
//  - 148 sequential steps; per step one fused kernel computes the four
//    512x512x512 gate GEMMs with v_wmma_f32_16x16x32_bf16 (f32 accumulate),
//    then applies the cell update in-register and writes h (f32 + bf16) / cs.
//  - LDS is double-buffered and software-pipelined: while chunk k feeds the
//    WMMAs, chunk k+1's weight tiles stream in via
//    GLOBAL_LOAD_ASYNC_TO_LDS_B128 (ASYNCcnt) and h-tile loads are in flight.
//  - bf16 h is double-buffered across steps (inter-workgroup RAW hazard).
//  - out = V @ h @ TEST computed by a deterministic 1-WG reduction kernel,
//    launched only for steps whose output is consumed (t >= 127).
// ---------------------------------------------------------------------------

#define HDIM   512
#define T_SEQ  128
#define AR_N   20
#define NSTEP  (T_SEQ + AR_N)   // 148
#define TILE   64
#define KT     32
#define NCHUNK (HDIM / KT)      // 16
#define LSTRIDE 40              // padded LDS row stride (bf16 elems): 80B, 16B-aligned

typedef __attribute__((ext_vector_type(16))) __bf16 v16bf;
typedef __attribute__((ext_vector_type(8)))  float  v8f;
typedef unsigned short bh16;
typedef unsigned int   u32;
typedef unsigned long long u64;

__device__ __forceinline__ bh16 f32_to_bf16(float f) {
  u32 u = __float_as_uint(f);
  u += 0x7fffu + ((u >> 16) & 1u);          // round-to-nearest-even
  return (bh16)(u >> 16);
}

__device__ __forceinline__ float sigmoid_(float x) {
  return 1.0f / (1.0f + __expf(-x));
}

// ---------------------------------------------------------------------------
// One recurrent step: pre[g] = W[g] @ h + Wx[g]*x + b[g]; cell update.
// Grid: 64 WGs (8x8 tiles of 64x64). Block: 256 threads = 8 waves.
// ---------------------------------------------------------------------------
__global__ __launch_bounds__(256)
void lstm_step_kernel(const bh16* __restrict__ Wbf,     // [4][H][H] bf16 (f,i,o,A)
                      const bh16* __restrict__ h_in,    // [H][H] bf16
                      bh16*       __restrict__ h_out,   // [H][H] bf16
                      float*      __restrict__ h_f32,   // [H][H] f32
                      float*      __restrict__ cs,      // [H][H] f32
                      const float* __restrict__ fWx, const float* __restrict__ iWx,
                      const float* __restrict__ oWx, const float* __restrict__ aWx,
                      const float* __restrict__ fb,  const float* __restrict__ ib,
                      const float* __restrict__ ob,  const float* __restrict__ ab,
                      const float* __restrict__ seq,
                      const float* __restrict__ slots,
                      int t)
{
  __shared__ bh16  lds_W[2][4][TILE][LSTRIDE];   // double-buffered gate tiles (40 KB)
  __shared__ bh16  lds_Ht[2][TILE][LSTRIDE];     // double-buffered h^T tiles (10 KB)
  __shared__ float lds_wx[4][TILE];
  __shared__ float lds_b[4][TILE];

  const int tid  = threadIdx.x;
  const int bx   = blockIdx.x;
  const int m_wg = (bx >> 3) * TILE;
  const int n_wg = (bx & 7)  * TILE;

  const int wave = tid >> 5;                  // wave32
  const int lane = tid & 31;
  const int hh   = lane >> 4;                 // lane half (0/1)
  const int l16  = lane & 15;
  const int tr   = wave >> 1;                 // tile row 0..3 (16-row tiles)
  const int tc2  = (wave & 1) * 2;            // tile cols {0,1} or {2,3}

  const float x = (t < T_SEQ) ? seq[t] : slots[t - 1];

  // stage per-row broadcast terms for this WG's 64 rows
  {
    const float* wxp[4] = {fWx, iWx, oWx, aWx};
    const float* bp [4] = {fb,  ib,  ob,  ab};
    for (int u = tid; u < 4 * TILE; u += 256) {
      const int g = u >> 6, r = u & 63;
      lds_wx[g][r] = wxp[g][m_wg + r];
      lds_b [g][r] = bp [g][m_wg + r];
    }
  }

  // ---- staging helpers ----------------------------------------------------
  // 4 gate W tiles (16 KB/chunk): 16 B per lane per issue, straight to LDS.
  auto stage_W_async = [&](int buf, int k0) {
#pragma unroll
    for (int it = 0; it < 4; ++it) {
      const int u   = tid + it * 256;         // 0..1023 transfers of 16 B
      const int g   = u >> 8;                 // 0..3
      const int rem = u & 255;
      const int row = rem >> 2;               // 0..63
      const int c8  = (rem & 3) * 8;          // bf16 elems (16 B chunks)
      const bh16* gptr = Wbf + (size_t)g * HDIM * HDIM
                             + (size_t)(m_wg + row) * HDIM + k0 + c8;
      const unsigned lofs = (unsigned)(u64)(const void*)&lds_W[buf][g][row][c8];
      asm volatile("global_load_async_to_lds_b128 %0, %1, off"
                   :: "v"(lofs), "v"((u64)(const void*)gptr)
                   : "memory");
    }
  };
  // h tile (32x64): global loads into VGPRs ...
  u32 hv4[4];
  auto load_h = [&](int k0) {
#pragma unroll
    for (int it = 0; it < 4; ++it) {
      const int u  = tid + it * 256;          // 0..1023
      const int kl = u >> 5;                  // 0..31
      const int nl = (u & 31) * 2;
      hv4[it] = *(const u32*)(h_in + (size_t)(k0 + kl) * HDIM + n_wg + nl);
    }
  };
  // ... then transposed into LDS -> [n][k], K-contiguous for B fragments.
  auto store_h = [&](int buf) {
#pragma unroll
    for (int it = 0; it < 4; ++it) {
      const int u  = tid + it * 256;
      const int kl = u >> 5;
      const int nl = (u & 31) * 2;
      lds_Ht[buf][nl    ][kl] = (bh16)(hv4[it] & 0xffffu);
      lds_Ht[buf][nl + 1][kl] = (bh16)(hv4[it] >> 16);
    }
  };

  v8f acc[4][2];
  const v8f vzero = {0.f, 0.f, 0.f, 0.f, 0.f, 0.f, 0.f, 0.f};
#pragma unroll
  for (int g = 0; g < 4; ++g)
#pragma unroll
    for (int j = 0; j < 2; ++j) acc[g][j] = vzero;

  union Frag { uint4 q[2]; v16bf v; };

  // ---- pipeline prologue: stage chunk 0 into buffer 0 ---------------------
  stage_W_async(0, 0);
  load_h(0);
  store_h(0);
  asm volatile("s_wait_asynccnt 0x0" ::: "memory");
  __syncthreads();

  // ---- software-pipelined main loop ---------------------------------------
  for (int ck = 0; ck < NCHUNK; ++ck) {
    const int p     = ck & 1;
    const int knext = (ck + 1) * KT;

    // kick off next chunk's staging while this chunk computes
    if (knext < HDIM) {
      stage_W_async(p ^ 1, knext);
      load_h(knext);
      if (knext + KT < HDIM)
        __builtin_prefetch(h_in + (size_t)(knext + KT + (tid >> 5)) * HDIM + n_wg, 0, 1);
    }

    // ---- fragments from buffer p (all loads first, then 8 WMMAs) ----------
    // B (32x16): V0..3 K=2d..2d+1 (+8*half), V4..7 K=16+2(d-4) (+8*half)
    Frag bfr[2];
#pragma unroll
    for (int j = 0; j < 2; ++j) {
      const int nloc = (tc2 + j) * 16 + l16;
      bfr[j].q[0] = *(const uint4*)&lds_Ht[p][nloc][8 * hh];
      bfr[j].q[1] = *(const uint4*)&lds_Ht[p][nloc][16 + 8 * hh];
    }
    // A (16x32): V0..3 K=2d..2d+1 (+8*half), V4..7 K=16+2(d-4) (+8*half)
    Frag afr[4];
    const int mloc = tr * 16 + l16;
#pragma unroll
    for (int g = 0; g < 4; ++g) {
      afr[g].q[0] = *(const uint4*)&lds_W[p][g][mloc][8 * hh];
      afr[g].q[1] = *(const uint4*)&lds_W[p][g][mloc][16 + 8 * hh];
    }
#pragma unroll
    for (int g = 0; g < 4; ++g)
#pragma unroll
      for (int j = 0; j < 2; ++j)
        acc[g][j] = __builtin_amdgcn_wmma_f32_16x16x32_bf16(
            false, afr[g].v, false, bfr[j].v, (short)0, acc[g][j], false, false);

    // finish next chunk's h transpose into the other buffer
    if (knext < HDIM) store_h(p ^ 1);

    asm volatile("s_wait_asynccnt 0x0" ::: "memory");  // next W tiles landed
    __syncthreads();                                   // publish buffer p^1
  }

  // ---- fused cell update: C/D layout lane -> n = lane&15, m = vgpr + 8*half ----
#pragma unroll
  for (int j = 0; j < 2; ++j) {
    const int n_loc = (tc2 + j) * 16 + l16;
    const int n_g   = n_wg + n_loc;
#pragma unroll
    for (int v = 0; v < 8; ++v) {
      const int m_loc = tr * 16 + hh * 8 + v;
      const int m_g   = m_wg + m_loc;
      const float fpre = acc[0][j][v] + lds_wx[0][m_loc] * x + lds_b[0][m_loc];
      const float ipre = acc[1][j][v] + lds_wx[1][m_loc] * x + lds_b[1][m_loc];
      const float opre = acc[2][j][v] + lds_wx[2][m_loc] * x + lds_b[2][m_loc];
      const float cpre = acc[3][j][v] + lds_wx[3][m_loc] * x + lds_b[3][m_loc];
      const float fg = sigmoid_(fpre);
      const float ig = sigmoid_(ipre);
      const float og = sigmoid_(opre);
      const float ch = tanhf(cpre);
      const size_t idx = (size_t)m_g * HDIM + n_g;
      const float cn = fg * cs[idx] + ig * ch;
      cs[idx] = cn;
      const float hv = og * tanhf(cn);
      h_f32[idx] = hv;
      h_out[idx] = f32_to_bf16(hv);
    }
  }
}

// ---------------------------------------------------------------------------
// out = sum_n TEST[n] * sum_m V[m] * h[m,n]   (deterministic single-WG reduce)
// ---------------------------------------------------------------------------
__global__ __launch_bounds__(512)
void out_reduce_kernel(const float* __restrict__ h_f32,
                       const float* __restrict__ Vv,
                       const float* __restrict__ Tv,
                       float* __restrict__ slots,
                       float* __restrict__ dout,
                       int t)
{
  __shared__ float red[512];
  const int n = threadIdx.x;
  float r = 0.f;
  for (int m = 0; m < HDIM; ++m)
    r += Vv[m] * h_f32[(size_t)m * HDIM + n];
  red[n] = r * Tv[n];
  __syncthreads();
  for (int s = 256; s > 0; s >>= 1) {
    if (n < s) red[n] += red[n + s];
    __syncthreads();
  }
  if (n == 0) {
    slots[t] = red[0];
    if (t >= T_SEQ) dout[t - T_SEQ] = red[0];
  }
}

__global__ __launch_bounds__(256)
void convert_w_kernel(const float* __restrict__ src, bh16* __restrict__ dst, int n) {
  const int i = blockIdx.x * 256 + threadIdx.x;
  if (i < n) dst[i] = f32_to_bf16(src[i]);
}

__global__ __launch_bounds__(256)
void zero_state_kernel(bh16* __restrict__ hbf0, float* __restrict__ cs, int n) {
  const int i = blockIdx.x * 256 + threadIdx.x;
  if (i < n) { hbf0[i] = 0; cs[i] = 0.f; }
}

// ---------------------------------------------------------------------------
extern "C" void kernel_launch(void* const* d_in, const int* in_sizes, int n_in,
                              void* d_out, int out_size, void* d_ws, size_t ws_size,
                              hipStream_t stream)
{
  const float* fWx = (const float*)d_in[0];
  const float* fWh = (const float*)d_in[1];
  const float* fb  = (const float*)d_in[2];
  const float* iWx = (const float*)d_in[3];
  const float* iWh = (const float*)d_in[4];
  const float* ib  = (const float*)d_in[5];
  const float* oWx = (const float*)d_in[6];
  const float* oWh = (const float*)d_in[7];
  const float* ob  = (const float*)d_in[8];
  const float* aWx = (const float*)d_in[9];
  const float* aWh = (const float*)d_in[10];
  const float* ab  = (const float*)d_in[11];
  const float* Vv  = (const float*)d_in[13];
  const float* Tv  = (const float*)d_in[14];
  const float* seq = (const float*)d_in[15];
  float* out = (float*)d_out;

  // workspace layout (~5.25 MB)
  char* ws = (char*)d_ws;
  const size_t WHH = (size_t)HDIM * HDIM;           // 262144
  bh16*  Wbf  = (bh16*)(ws);                        // 4*WHH bf16 = 2 MB
  bh16*  hbf0 = (bh16*)(ws + 2097152);              // 512 KB
  bh16*  hbf1 = (bh16*)(ws + 2097152 + 524288);     // 512 KB
  float* hf32 = (float*)(ws + 3145728);             // 1 MB
  float* cs   = (float*)(ws + 3145728 + 1048576);   // 1 MB
  float* slots= (float*)(ws + 3145728 + 2097152);   // 148 f32

  const int nb = (int)(WHH / 256);
  convert_w_kernel<<<nb, 256, 0, stream>>>(fWh, Wbf + 0 * WHH, (int)WHH);
  convert_w_kernel<<<nb, 256, 0, stream>>>(iWh, Wbf + 1 * WHH, (int)WHH);
  convert_w_kernel<<<nb, 256, 0, stream>>>(oWh, Wbf + 2 * WHH, (int)WHH);
  convert_w_kernel<<<nb, 256, 0, stream>>>(aWh, Wbf + 3 * WHH, (int)WHH);
  zero_state_kernel<<<nb, 256, 0, stream>>>(hbf0, cs, (int)WHH);

  for (int t = 0; t < NSTEP; ++t) {
    const bh16* hin  = (t & 1) ? hbf1 : hbf0;
    bh16*       hout = (t & 1) ? hbf0 : hbf1;
    lstm_step_kernel<<<64, 256, 0, stream>>>(Wbf, hin, hout, hf32, cs,
                                             fWx, iWx, oWx, aWx,
                                             fb, ib, ob, ab,
                                             seq, slots, t);
    if (t >= T_SEQ - 1)   // slot t=127 feeds first AR step; t>=128 are outputs
      out_reduce_kernel<<<1, 512, 0, stream>>>(hf32, Vv, Tv, slots, out, t);
  }
}
